// EBCNN_79413945303391
// MI455X (gfx1250) — matrix operand order repaired and checked by reading, not compiled, not run on hardware
//
#include <hip/hip_runtime.h>
#include <hip/hip_fp16.h>
#include <math.h>

typedef __attribute__((ext_vector_type(16))) _Float16 v16h;
typedef __attribute__((ext_vector_type(8)))  _Float16 v8h;
typedef __attribute__((ext_vector_type(8)))  float    v8f;

#define S_LEN 30
#define D_CH  512
#define KDIM  1536          // 3*D feature length (K of the GEMM)
#define K8    (KDIM / 8)    // 192 16-byte chunks per feature row
#define ROWP  1544          // LDS row pitch in f16 (+16B pad -> conflict-free)

// Single-instruction lane-XOR exchange: ds_swizzle_b32 group-of-32 mode,
// offset = {1'b0, xor_mask[14:10], or_mask=0, and_mask=0x1F}.
#define SWIZZLE_XOR_ADD(v, mask)                                              \
    (v) += __int_as_float(__builtin_amdgcn_ds_swizzle(                        \
               __float_as_int(v), (0x1F | ((mask) << 10))))

// Branchless tanh: native v_tanh_f32 on gfx1250 if the builtin is declared,
// else a divergence-free exp-based formulation (avoids the branchy OCML path).
__device__ __forceinline__ float fast_tanh(float x) {
#if defined(__has_builtin)
#if __has_builtin(__builtin_amdgcn_tanhf)
    return __builtin_amdgcn_tanhf(x);
#else
    float t = fminf(fmaxf(x, -15.0f), 15.0f);
    float e = __expf(2.0f * t);
    return (e - 1.0f) / (e + 1.0f);
#endif
#else
    float t = fminf(fmaxf(x, -15.0f), 15.0f);
    float e = __expf(2.0f * t);
    return (e - 1.0f) / (e + 1.0f);
#endif
}

// ---------------------------------------------------------------- zero out
__global__ void zero_out_kernel(float* __restrict__ out, int n) {
    int i = blockIdx.x * blockDim.x + threadIdx.x;
    if (i < n) out[i] = 0.0f;
}

// ---------------------------------------------------------------- W1 -> f16
__global__ void convert_w1_kernel(const float* __restrict__ w1,
                                  _Float16* __restrict__ w1h, int n) {
    int i = blockIdx.x * blockDim.x + threadIdx.x;
    int stride = gridDim.x * blockDim.x;
    for (; i < n; i += stride) w1h[i] = (_Float16)w1[i];
}

// ---------------------------------------------------------------- features
// One block per batch row b; thread c handles channel c.
// vl = max_i conv3(x, w_long), i=0..27 ; vm = max over i=23..27 with w_mid ;
// vs = x[29]. Written as an f16 row [vl(512) | vm(512) | vs(512)].
__global__ __launch_bounds__(D_CH)
void feature_kernel(const float* __restrict__ emb,
                    const float* __restrict__ wl,
                    const float* __restrict__ wm,
                    _Float16* __restrict__ feat) {
    const int b = blockIdx.x;
    const int c = threadIdx.x;
    const float wl0 = wl[0], wl1 = wl[1], wl2 = wl[2];
    const float wm0 = wm[0], wm1 = wm[1], wm2 = wm[2];

    const float* xb = emb + (size_t)b * (S_LEN * D_CH) + c;
    float x0 = xb[0 * D_CH];
    float x1 = xb[1 * D_CH];
    float x2 = 0.0f;
    float vl = -INFINITY, vm = -INFINITY;

    #pragma unroll
    for (int s = 2; s < S_LEN; ++s) {
        x2 = xb[s * D_CH];
        float cl = x0 * wl0 + x1 * wl1 + x2 * wl2;   // conv index i = s-2
        vl = fmaxf(vl, cl);
        if (s >= 25) {                               // i = 23..27 (last-7 window)
            float cm = x0 * wm0 + x1 * wm1 + x2 * wm2;
            vm = fmaxf(vm, cm);
        }
        x0 = x1; x1 = x2;
    }

    _Float16* fr = feat + (size_t)b * KDIM;
    fr[c]            = (_Float16)vl;
    fr[D_CH + c]     = (_Float16)vm;
    fr[2 * D_CH + c] = (_Float16)x2;                 // x2 == emb[b, 29, c]
}

// ---------------------------------------------------------------- fused MLP
// Block: 256 threads (8 wave32). blockIdx.x = M-tile (16 batch rows),
// blockIdx.y = H-tile (128 columns, 16 per wave).
// hidden tile computed via v_wmma_f32_16x16x32_f16, K = 1536 (48 steps),
// then tanh -> *W2[n] -> 16-lane swizzle reduce -> atomicAdd out[b].
__global__ __launch_bounds__(256)
void mlp_kernel(const _Float16* __restrict__ feat,
                const _Float16* __restrict__ w1h,
                const float*    __restrict__ w2,
                float*          __restrict__ out) {
    __shared__ __align__(16) _Float16 atile[16 * ROWP];   // 49,408 B

    const int tid   = threadIdx.x;
    const int mbase = blockIdx.x * 16;

    // Cooperative A-tile stage: 16 rows x 1536 f16 = 3072 x 16B chunks.
    {
        const uint4* src = (const uint4*)(feat + (size_t)mbase * KDIM);
        #pragma unroll
        for (int t = 0; t < 12; ++t) {
            int ci  = tid + t * 256;
            int row = ci / K8;
            int col = ci % K8;
            uint4 v = src[row * K8 + col];
            *(uint4*)(&atile[row * ROWP + col * 8]) = v;
        }
    }
    __syncthreads();

    const int lane = tid & 31;
    const int wid  = tid >> 5;
    const int nl   = lane & 15;       // A: row M ; B: column N ; C/D: N
    const int ks   = lane >> 4;       // K-half selector per ISA layouts
    const int nbase = blockIdx.y * 128 + wid * 16;

    const _Float16* arow = atile + nl * ROWP;                    // M = lane%16
    const _Float16* brow = w1h + (size_t)(nbase + nl) * KDIM;    // N = lane%16

    v8f acc = {};
    #pragma unroll 4
    for (int kb = 0; kb < KDIM; kb += 32) {
        // A 16x32 f16: lanes 0-15 hold K {0..7,16..23}, lanes 16-31 {8..15,24..31}
        v8h alo = *(const v8h*)(arow + kb + ks * 8);
        v8h ahi = *(const v8h*)(arow + kb + 16 + ks * 8);
        v16h a  = __builtin_shufflevector(alo, ahi,
                    0,1,2,3,4,5,6,7,8,9,10,11,12,13,14,15);
        // B 32x16 f16: lane = N, same K order as A: {0..7,16..23} / {8..15,24..31}
        v8h blo = *(const v8h*)(brow + kb + ks * 8);
        v8h bhi = *(const v8h*)(brow + kb + 16 + ks * 8);
        v16h b  = __builtin_shufflevector(blo, bhi,
                    0,1,2,3,4,5,6,7,8,9,10,11,12,13,14,15);
        acc = __builtin_amdgcn_wmma_f32_16x16x32_f16(
                  false, a, false, b, (short)0, acc, false, false);
    }

    const float w2v = w2[nbase + nl];

    // C/D layout: VGPR r, lane L -> (M = r + 8*(L/16), N = L%16)
    #pragma unroll
    for (int r = 0; r < 8; ++r) {
        float contrib = fast_tanh(acc[r]) * w2v;
        SWIZZLE_XOR_ADD(contrib, 1);   // butterfly over the 16 N-lanes
        SWIZZLE_XOR_ADD(contrib, 2);
        SWIZZLE_XOR_ADD(contrib, 4);
        SWIZZLE_XOR_ADD(contrib, 8);
        if (nl == 0) atomicAdd(&out[mbase + r + 8 * ks], contrib);
    }
}

// ---------------------------------------------------------------- launcher
extern "C" void kernel_launch(void* const* d_in, const int* in_sizes, int n_in,
                              void* d_out, int out_size, void* d_ws, size_t ws_size,
                              hipStream_t stream) {
    const float* emb = (const float*)d_in[0];   // [B, 30, 512]
    const float* wl  = (const float*)d_in[1];   // [3]
    const float* wm  = (const float*)d_in[2];   // [3]
    const float* W1  = (const float*)d_in[3];   // [H, 1536]
    const float* W2  = (const float*)d_in[4];   // [1, H]
    float* out = (float*)d_out;                 // [B]

    const int B = in_sizes[0] / (S_LEN * D_CH); // 8192
    const int H = in_sizes[4];                  // 1024

    // Workspace: f16 feature [B,1536] then f16 W1 [H,1536]
    _Float16* featH = (_Float16*)d_ws;
    _Float16* w1H   = (_Float16*)((char*)d_ws + (size_t)B * KDIM * sizeof(_Float16));

    zero_out_kernel<<<(out_size + 255) / 256, 256, 0, stream>>>(out, out_size);
    convert_w1_kernel<<<256, 256, 0, stream>>>(W1, w1H, H * KDIM);
    feature_kernel<<<B, D_CH, 0, stream>>>(emb, wl, wm, featH);

    dim3 grid(B / 16, H / 128);
    mlp_kernel<<<grid, 256, 0, stream>>>(featH, w1H, W2, out);
}
